// WeightedCoxHazardLoss_33595234189715
// MI455X (gfx1250) — matrix operand order repaired and checked by reading, not compiled, not run on hardware
//
#include <hip/hip_runtime.h>
#include <hip/hip_bf16.h>
#include <math.h>

typedef __attribute__((ext_vector_type(16))) _Float16 v16h;
typedef __attribute__((ext_vector_type(8)))  float    v8f;

#define NN 256
#define EPSF 1e-07f

// One block (256 threads = 8 wave32) per batch row.
__global__ __launch_bounds__(256) void cox_batch_kernel(
    const float* __restrict__ pred,
    const float* __restrict__ target,
    const int*   __restrict__ valid,
    float* __restrict__ ws, int B)
{
    const int b = blockIdx.x;
    const int t = threadIdx.x;

    __shared__ float    s_tm[NN];     // masked target
    __shared__ float    s_e[NN];      // exp(pred - gm), 0 for invalid
    __shared__ _Float16 s_eh[NN];     // f16 copy for WMMA B operand
    __shared__ float    s_S[NN];      // risk-set exp sums
    __shared__ float    s_red[256];   // reduction scratch

    const int   v  = valid[(size_t)b * NN + t];
    const float p  = pred [(size_t)b * NN + t];
    const float tg = target[(size_t)b * NN + t];
    const float tm = v ? tg : -1.0f;
    s_tm[t] = tm;
    __syncthreads();

    // ---- block reductions: bmax = max(tm); gm = max(valid pred); nvalid ----
    s_red[t] = tm;
    __syncthreads();
    for (int off = 128; off > 0; off >>= 1) {
        if (t < off) s_red[t] = fmaxf(s_red[t], s_red[t + off]);
        __syncthreads();
    }
    const float bmax = s_red[0];
    __syncthreads();

    s_red[t] = v ? p : -INFINITY;
    __syncthreads();
    for (int off = 128; off > 0; off >>= 1) {
        if (t < off) s_red[t] = fmaxf(s_red[t], s_red[t + off]);
        __syncthreads();
    }
    const float gm = s_red[0];
    __syncthreads();

    s_red[t] = v ? 1.0f : 0.0f;
    __syncthreads();
    for (int off = 128; off > 0; off >>= 1) {
        if (t < off) s_red[t] += s_red[t + off];
        __syncthreads();
    }
    const float nvalid = s_red[0];
    __syncthreads();

    // ---- stable exp terms ----
    const float e = v ? __expf(p - gm) : 0.0f;
    s_e[t]  = e;
    s_eh[t] = (_Float16)e;
    __syncthreads();

    // ---- S[i] = sum_j (tm[j] >= tm[i]) * e[j]  via V_WMMA_F32_16X16X32_F16 ----
    // Wave w handles row tiles i0 = 16*(w) and 16*(w+8). A = risk tile (0/1 f16),
    // B = e[j] broadcast across all 16 columns, C accumulates in f32.
    {
        const int lane = t & 31;
        const int w    = t >> 5;
        const int h    = lane >> 4;   // half-wave: selects K-group / M-group
        const int m    = lane & 15;   // row within A tile / column of B,D

        for (int ii = 0; ii < 2; ++ii) {              // uniform across block
            const int   i0  = 16 * (w + 8 * ii);
            const float tmi = s_tm[i0 + m];
            v8f c = {};
            for (int j0 = 0; j0 < NN; j0 += 32) {     // uniform K loop
                v16h a, bb;
                #pragma unroll
                for (int vv = 0; vv < 8; ++vv) {
                    // A (16x32 f16): VGPR vv<4 holds K=8h+2vv{,+1}; vv>=4 holds K=16+8h+2(vv-4){,+1}
                    const int ka = (vv < 4) ? (8 * h + 2 * vv)
                                            : (16 + 8 * h + 2 * (vv - 4));
                    a[2 * vv]     = (s_tm[j0 + ka]     >= tmi) ? (_Float16)1.0f : (_Float16)0.0f;
                    a[2 * vv + 1] = (s_tm[j0 + ka + 1] >= tmi) ? (_Float16)1.0f : (_Float16)0.0f;
                    // B (32x16 f16): VGPR vv holds same K as A (table order) at N = m; columns identical
                    const int kb = ka;
                    bb[2 * vv]     = s_eh[j0 + kb];
                    bb[2 * vv + 1] = s_eh[j0 + kb + 1];
                }
                c = __builtin_amdgcn_wmma_f32_16x16x32_f16(
                        false, a, false, bb, (short)0, c, false, false);
            }
            // D layout: c[r] = D[M = r + 8h][N = m]; all N columns equal.
            if (m == 0) {
                #pragma unroll
                for (int r = 0; r < 8; ++r) s_S[i0 + 8 * h + r] = c[r];
            }
        }
    }
    __syncthreads();

    // ---- per-item loss: part1 + part2 ----
    const float S        = s_S[t];
    const float log_den  = gm + __logf(S);
    const float bmax_sf  = fmaxf(bmax, 1.0f);
    const bool  is_elim  = (tm < bmax) && (tm > 0.0f) && (v != 0);
    const float wgt      = fminf(fmaxf((bmax - tm) / bmax_sf, 0.0f), 1.0f);
    const float we       = is_elim ? wgt : 0.0f;

    float loss_i = 0.0f;
    if (we > 0.0f) {
        const float inv = 1.0f / S;          // probs[i,j] = e[j] / S[i]
        float l2 = 0.0f;
        #pragma unroll 4
        for (int j = 0; j < NN; ++j) {
            const float ej = s_e[j];
            if ((s_tm[j] >= tm) && (ej > 0.0f) && (j != t)) {
                float pr = ej * inv;
                pr = fminf(pr, 1.0f - EPSF);
                l2 -= __logf(1.0f - pr + EPSF);
            }
        }
        loss_i = we * ((log_den - p) + l2);  // part1 = -(pred - log_den)*we
    }

    // ---- block sum -> per-batch totals ----
    s_red[t] = loss_i;
    __syncthreads();
    for (int off = 128; off > 0; off >>= 1) {
        if (t < off) s_red[t] += s_red[t + off];
        __syncthreads();
    }
    if (t == 0) {
        const float vb = (nvalid >= 2.0f) ? 1.0f : 0.0f;
        ws[b]     = s_red[0] * vb;
        ws[B + b] = vb;
    }
}

__global__ __launch_bounds__(256) void cox_final_kernel(
    const float* __restrict__ ws, float* __restrict__ out, int B)
{
    __shared__ float s1[256];
    __shared__ float s2[256];
    const int t = threadIdx.x;
    float a = 0.0f, c = 0.0f;
    for (int i = t; i < B; i += 256) { a += ws[i]; c += ws[B + i]; }
    s1[t] = a; s2[t] = c;
    __syncthreads();
    for (int off = 128; off > 0; off >>= 1) {
        if (t < off) { s1[t] += s1[t + off]; s2[t] += s2[t + off]; }
        __syncthreads();
    }
    if (t == 0) out[0] = s1[0] / fmaxf(s2[0], 1.0f);
}

extern "C" void kernel_launch(void* const* d_in, const int* in_sizes, int n_in,
                              void* d_out, int out_size, void* d_ws, size_t ws_size,
                              hipStream_t stream)
{
    const float* pred   = (const float*)d_in[0];
    const float* target = (const float*)d_in[1];
    const int*   valid  = (const int*)d_in[2];
    float* out = (float*)d_out;
    float* ws  = (float*)d_ws;

    const int B = in_sizes[0] / NN;

    cox_batch_kernel<<<B, 256, 0, stream>>>(pred, target, valid, ws, B);
    cox_final_kernel<<<1, 256, 0, stream>>>(ws, out, B);
}
